// QKVAttentionLegacy_86062554677918
// MI455X (gfx1250) — matrix operand order repaired and checked by reading, not compiled, not run on hardware
//
#include <hip/hip_runtime.h>
#include <stdint.h>

typedef __attribute__((ext_vector_type(16))) _Float16 v16h;
typedef __attribute__((ext_vector_type(8)))  _Float16 v8h;
typedef __attribute__((ext_vector_type(2)))  _Float16 v2h;
typedef __attribute__((ext_vector_type(8)))  float    v8f;
typedef __attribute__((ext_vector_type(8)))  uint32_t v8u;

// qkv: (8, 1536, 2048) fp32. per (batch,head): q rows c=0..63, k rows 64..127,
// v rows 128..191; each row = 2048 contiguous floats over position.
// out: (8, 512, 2048) fp32.
//
// Flash attention, f16 WMMA math / f32 online softmax (log2-domain, raw
// v_exp_f32 via __builtin_amdgcn_exp2f -- no libm range fixup).
// Block = 256 threads = 8 waves, one head, 8 adjacent 16-query tiles.
// Per 32-key chunk the block stages K^T and V (f32 -> f16 once) into
// double-buffered LDS; next chunk's global loads are issued before the
// current chunk's convert/store/compute so they stay in flight.

#define QK_SCALE 0.125f          // (1/sqrt(sqrt(64)))^2 folded into Q
#define LOG2E    1.4426950408889634f

// LDS layout (per buffer, element = f16):
//   K^T tile: [32 s][72 pitch]  (36 dwords/row -> 16 b128 lanes tile 64 banks)
//   V  tile : [64 c][40 pitch]  (20 dwords/row -> 16 b128 lanes tile 64 banks)
#define KTP   72
#define VP    40
#define KT_SZ (32 * KTP)               // 2304 f16
#define BUF_SZ (KT_SZ + 64 * VP)       // 4864 f16 = 9728 B

struct Stage { float k[8]; float4 va, vb; };

static __device__ __forceinline__ uint32_t pk_f16x2(float lo, float hi) {
    v2h t; t[0] = (_Float16)lo; t[1] = (_Float16)hi;    // v_cvt_pk_f16_f32
    return __builtin_bit_cast(uint32_t, t);
}

// raw hardware exp2 (single v_exp_f32, no denormal-range guard).
static __device__ __forceinline__ float fast_exp2(float x) {
    return __builtin_amdgcn_exp2f(x);
}

__global__ __launch_bounds__(256)
void flash_attn_qkv_kernel(const float* __restrict__ qkv, float* __restrict__ out)
{
    constexpr int LEN = 2048;
    constexpr int CH  = 64;
    constexpr int NIT = LEN / 32;

    __shared__ alignas(16) _Float16 smem[2 * BUF_SZ];

    const int tid  = threadIdx.x;
    const int lane = tid & 31;
    const int ln   = lane & 15;          // N index (t) / A-row index
    const int lh   = lane >> 4;          // lane half
    const int wave = tid >> 5;

    const int ttile = blockIdx.x * 8 + wave;   // 0..127
    const int tbase = ttile * 16;
    const int b     = blockIdx.y;              // 0..63
    const int batch = b >> 3, head = b & 7;

    const float* base = qkv + (size_t)(batch * 1536 + head * 192) * LEN;
    const float* qp = base;
    const float* kp = base + (size_t)CH * LEN;
    const float* vp = base + (size_t)(2 * CH) * LEN;

    // staging assignments (per thread, per chunk)
    const int ks  = tid >> 3;            // K^T: s row 0..31
    const int kc0 = (tid & 7) * 8;       //       c run start
    const int vc  = tid >> 2;            // V:   c row 0..63
    const int vs0 = (tid & 3) * 8;       //       s run start

    // ---- Q as two B-fragments (K = c within 32-chunk, N = t), scale folded ----
    v16h Bq[2];
    #pragma unroll
    for (int h = 0; h < 2; ++h) {
        #pragma unroll
        for (int j = 0; j < 16; ++j) {
            const int c = h * 32 + (j & 7) + 8 * lh + 16 * (j >> 3);   // B elem j -> K = (j&7) + 8*lh + 16*(j>>3)
            Bq[h][j] = (_Float16)(QK_SCALE * qp[(size_t)c * LEN + tbase + ln]);
        }
    }

    v8f o[4];
    #pragma unroll
    for (int ct = 0; ct < 4; ++ct) o[ct] = (v8f){};
    float m_run = -3.0e38f;              // running max in log2 units
    float l_run = 0.0f;

    // ---- software-pipelined staging loads ----
    auto stage_load = [&](int sb) {
        Stage st;
        #pragma unroll
        for (int i = 0; i < 8; ++i)
            st.k[i] = kp[(size_t)(kc0 + i) * LEN + sb + ks];
        st.va = *(const float4*)(vp + (size_t)vc * LEN + sb + vs0);
        st.vb = *(const float4*)(vp + (size_t)vc * LEN + sb + vs0 + 4);
        return st;
    };

    Stage cur = stage_load(0);

    for (int it = 0; it < NIT; ++it) {
        _Float16* lds = smem + (it & 1) * BUF_SZ;

        // issue next chunk's global loads first: in flight through barrier+compute
        Stage nxt = cur;
        if (it + 1 < NIT) nxt = stage_load((it + 1) * 32);

        // ---- convert + store current chunk ----
        {
            v8h kk;
            #pragma unroll
            for (int i = 0; i < 8; ++i) kk[i] = (_Float16)cur.k[i];
            *(v8h*)&lds[ks * KTP + kc0] = kk;

            v8h vv;
            vv[0] = (_Float16)cur.va.x; vv[1] = (_Float16)cur.va.y;
            vv[2] = (_Float16)cur.va.z; vv[3] = (_Float16)cur.va.w;
            vv[4] = (_Float16)cur.vb.x; vv[5] = (_Float16)cur.vb.y;
            vv[6] = (_Float16)cur.vb.z; vv[7] = (_Float16)cur.vb.w;
            *(v8h*)&lds[KT_SZ + vc * VP + vs0] = vv;
        }
        __syncthreads();   // one barrier/chunk; double buffering covers reuse

        // ---- logits^T = K^T x Q : ct0 rows s=+0..15, ct1 rows s=+16..31 ----
        v8f ct0 = (v8f){}, ct1 = (v8f){};
        #pragma unroll
        for (int h = 0; h < 2; ++h) {
            const v8h a0lo = *(const v8h*)&lds[(0 + ln) * KTP + h * 32 + 8 * lh];
            const v8h a0hi = *(const v8h*)&lds[(0 + ln) * KTP + h * 32 + 16 + 8 * lh];
            const v8h a1lo = *(const v8h*)&lds[(16 + ln) * KTP + h * 32 + 8 * lh];
            const v8h a1hi = *(const v8h*)&lds[(16 + ln) * KTP + h * 32 + 16 + 8 * lh];
            const v16h a0 = __builtin_shufflevector(a0lo, a0hi,
                0,1,2,3,4,5,6,7,8,9,10,11,12,13,14,15);
            const v16h a1 = __builtin_shufflevector(a1lo, a1hi,
                0,1,2,3,4,5,6,7,8,9,10,11,12,13,14,15);
            ct0 = __builtin_amdgcn_wmma_f32_16x16x32_f16(false, a0, false, Bq[h],
                                                         (short)0, ct0, false, false);
            ct1 = __builtin_amdgcn_wmma_f32_16x16x32_f16(false, a1, false, Bq[h],
                                                         (short)0, ct1, false, false);
        }

        // ---- online softmax over s (fragment rows), log2 domain ----
        float mx = ct0[0];
        #pragma unroll
        for (int r = 0; r < 8; ++r) { mx = fmaxf(mx, ct0[r]); mx = fmaxf(mx, ct1[r]); }
        mx = fmaxf(mx, __shfl_xor(mx, 16, 32));
        const float mn   = fmaxf(m_run, mx * LOG2E);
        const float corr = fast_exp2(m_run - mn);

        float p0[8], p1[8], sum = 0.0f;
        #pragma unroll
        for (int r = 0; r < 8; ++r) {
            p0[r] = fast_exp2(fmaf(ct0[r], LOG2E, -mn));   // v_(pk_)fma + v_exp
            p1[r] = fast_exp2(fmaf(ct1[r], LOG2E, -mn));
            sum  += p0[r] + p1[r];
        }
        sum  += __shfl_xor(sum, 16, 32);
        l_run = l_run * corr + sum;
        m_run = mn;
        #pragma unroll
        for (int ct = 0; ct < 4; ++ct)
            #pragma unroll
            for (int r = 0; r < 8; ++r) o[ct][r] *= corr;  // per-lane scalar (t-only)

        // ---- P^T -> B fragment: pack pairs first, then one xor16 per dword ----
        v8u bp;
        #pragma unroll
        for (int w = 0; w < 4; ++w) {
            const uint32_t a0 = pk_f16x2(p0[2 * w], p0[2 * w + 1]);
            const uint32_t a1 = pk_f16x2(p1[2 * w], p1[2 * w + 1]);
            const uint32_t x1 = (uint32_t)__shfl_xor((int)a1, 16, 32);
            const uint32_t x0 = (uint32_t)__shfl_xor((int)a0, 16, 32);
            bp[w]     = lh ? x1 : a0;   // K pairs (2w, 2w+1)       [+16 on half 1]
            bp[w + 4] = lh ? a1 : x0;   // K pairs (8+2w, 8+2w+1)   [+16 on half 1]
        }
        const v16h Bp = __builtin_bit_cast(v16h, bp);

        // ---- O^T += V x P^T ----
        #pragma unroll
        for (int ct = 0; ct < 4; ++ct) {
            const v8h vlo = *(const v8h*)&lds[KT_SZ + (ct * 16 + ln) * VP + 16 * lh];
            const v8h vhi = *(const v8h*)&lds[KT_SZ + (ct * 16 + ln) * VP + 16 * lh + 8];
            const v16h av = __builtin_shufflevector(vlo, vhi,
                0,1,2,3,4,5,6,7,8,9,10,11,12,13,14,15);
            o[ct] = __builtin_amdgcn_wmma_f32_16x16x32_f16(false, av, false, Bp,
                                                           (short)0, o[ct], false, false);
        }

        cur = nxt;
    }

    // ---- normalize and write O (fp32) ----
    const float inv_l = 1.0f / l_run;
    float* ob = out + (size_t)(batch * 512 + head * 64) * LEN;
    #pragma unroll
    for (int ct = 0; ct < 4; ++ct) {
        #pragma unroll
        for (int r = 0; r < 8; ++r) {
            const int c = ct * 16 + r + 8 * lh;
            ob[(size_t)c * LEN + tbase + ln] = o[ct][r] * inv_l;
        }
    }
}

extern "C" void kernel_launch(void* const* d_in, const int* in_sizes, int n_in,
                              void* d_out, int out_size, void* d_ws, size_t ws_size,
                              hipStream_t stream) {
    (void)in_sizes; (void)n_in; (void)d_ws; (void)ws_size; (void)out_size;
    const float* qkv = (const float*)d_in[0];
    float* out = (float*)d_out;
    // 128 query tiles of 16 rows; 8 waves per block -> grid.x = 16; grid.y = 64 heads
    dim3 grid(2048 / 16 / 8, 64);
    flash_attn_qkv_kernel<<<grid, dim3(256), 0, stream>>>(qkv, out);
}
